// gasGCNConv_52871047413953
// MI455X (gfx1250) — matrix operand order repaired and checked by reading, not compiled, not run on hardware
//
#include <hip/hip_runtime.h>
#include <hip/hip_bf16.h>

typedef __attribute__((ext_vector_type(16))) _Float16 v16h;
typedef __attribute__((ext_vector_type(8)))  _Float16 v8h;
typedef __attribute__((ext_vector_type(4)))  _Float16 v4h;
typedef __attribute__((ext_vector_type(8)))  float    v8f;

struct alignas(16) F4 { float v[4]; };

#define IN_CH   1024
#define OUT_CH  256

// ---------------------------------------------------------------------------
// Kernel 0: one-shot split of W into f16 hi/lo planes (row-major [256][1024]).
// ---------------------------------------------------------------------------
__global__ __launch_bounds__(256) void gcn_w_split(
    const float* __restrict__ W, _Float16* __restrict__ Whi,
    _Float16* __restrict__ Wlo, int n_f4)
{
    const int i4 = blockIdx.x * 256 + threadIdx.x;
    if (i4 < n_f4) {
        F4 w = reinterpret_cast<const F4*>(W)[i4];
        v4h hh, ll;
#pragma unroll
        for (int j = 0; j < 4; ++j) {
            _Float16 h = (_Float16)w.v[j];
            hh[j] = h;
            ll[j] = (_Float16)(w.v[j] - (float)h);
        }
        *reinterpret_cast<v4h*>(Whi + (size_t)i4 * 4) = hh;
        *reinterpret_cast<v4h*>(Wlo + (size_t)i4 * 4) = ll;
    }
}

// ---------------------------------------------------------------------------
// Kernel 1: h = X @ W^T + b   via v_wmma_f32_16x16x32_f16 (2-term f16 split).
// Block: 512 threads = 16 waves; wave w owns output channels [16w, 16w+16).
// Block covers 64 rows of X (4 M-tiles). X is split into f16 hi/lo ONCE while
// staging to LDS, stored in WMMA A-fragment order -> inner loop is pure
// ds_load_b128 + global_load_b128 + v_wmma.
// ---------------------------------------------------------------------------
__global__ __launch_bounds__(512) void gcn_gemm_wmma(
    const float* __restrict__ X, const _Float16* __restrict__ Whi,
    const _Float16* __restrict__ Wlo, const float* __restrict__ bias,
    float* __restrict__ H, int nrows)
{
    constexpr int MT = 4;                  // M-tiles per block
    // Fragment-order LDS: per row: [khalf(2)][pos(16)] halves, padded to 40.
    __shared__ _Float16 ldsH[64 * 40];
    __shared__ _Float16 ldsL[64 * 40];

    const int tid   = threadIdx.x;
    const int wave  = tid >> 5;            // 0..15 -> N-tile
    const int lane  = tid & 31;
    const int mloc  = lane & 15;
    const int khalf = lane >> 4;

    const int rbase = blockIdx.x * 64;
    const int ncol  = wave * 16 + mloc;    // output channel (row of W)

    v8f acc[MT];
    {
        v8f z;
#pragma unroll
        for (int i = 0; i < 8; ++i) z[i] = 0.0f;
#pragma unroll
        for (int t = 0; t < MT; ++t) acc[t] = z;
    }

    // staging thread's fixed coordinates
    const int srow = tid >> 3;             // 0..63
    const int scg  = (tid & 7) * 4;        // local k base: 0,4,...,28
    const int skh  = scg >> 4;             // fragment khalf of this k-quad
    const int spos = scg & 15;
    const int sidx = srow * 40 + skh * 16 + spos;
    const int sgr  = rbase + srow;

    for (int k0 = 0; k0 < IN_CH; k0 += 32) {
        __syncthreads();  // previous iteration's fragment reads done
        // ---- stage + split X[rbase..rbase+64) x [k0..k0+32) into LDS hi/lo
        {
            F4 x; x.v[0] = x.v[1] = x.v[2] = x.v[3] = 0.0f;
            if (sgr < nrows) {
                x = *reinterpret_cast<const F4*>(X + (size_t)sgr * IN_CH + k0 + scg);
                if (k0 + 32 < IN_CH)
                    __builtin_prefetch(X + (size_t)sgr * IN_CH + k0 + 32 + scg, 0, 1);
            }
            v4h hh, ll;
#pragma unroll
            for (int j = 0; j < 4; ++j) {
                _Float16 h = (_Float16)x.v[j];
                hh[j] = h;
                ll[j] = (_Float16)(x.v[j] - (float)h);
            }
            *reinterpret_cast<v4h*>(&ldsH[sidx]) = hh;
            *reinterpret_cast<v4h*>(&ldsL[sidx]) = ll;
        }
        __syncthreads();

        // ---- B fragments: 16 contiguous halves = 2x global_load_b128 each
        const size_t boff = (size_t)ncol * IN_CH + k0 + khalf * 16;
        v16h bh = *reinterpret_cast<const v16h*>(Whi + boff);
        v16h bl = *reinterpret_cast<const v16h*>(Wlo + boff);

        // ---- per M-tile: A fragments straight from LDS, 3 WMMAs
#pragma unroll
        for (int t = 0; t < MT; ++t) {
            const int abase = (t * 16 + mloc) * 40 + khalf * 16;
            v8h h0 = *reinterpret_cast<const v8h*>(&ldsH[abase]);
            v8h h1 = *reinterpret_cast<const v8h*>(&ldsH[abase + 8]);
            v8h l0 = *reinterpret_cast<const v8h*>(&ldsL[abase]);
            v8h l1 = *reinterpret_cast<const v8h*>(&ldsL[abase + 8]);
            v16h ah = __builtin_shufflevector(h0, h1, 0,1,2,3,4,5,6,7,8,9,10,11,12,13,14,15);
            v16h al = __builtin_shufflevector(l0, l1, 0,1,2,3,4,5,6,7,8,9,10,11,12,13,14,15);
            acc[t] = __builtin_amdgcn_wmma_f32_16x16x32_f16(
                false, ah, false, bh, (short)0, acc[t], false, false);
            acc[t] = __builtin_amdgcn_wmma_f32_16x16x32_f16(
                false, al, false, bh, (short)0, acc[t], false, false);
            acc[t] = __builtin_amdgcn_wmma_f32_16x16x32_f16(
                false, ah, false, bl, (short)0, acc[t], false, false);
        }
    }

    // ---- epilogue: D layout -> lane = N, VGPR r -> M = r + 8*khalf
    const float bv = bias[ncol];
#pragma unroll
    for (int t = 0; t < MT; ++t) {
#pragma unroll
        for (int r = 0; r < 8; ++r) {
            const int grow = rbase + t * 16 + r + khalf * 8;
            if (grow < nrows)
                H[(size_t)grow * OUT_CH + ncol] = acc[t][r] + bv;
        }
    }
}

// ---------------------------------------------------------------------------
// Kernel 2: out = self_edge_weight[:,None] * h     (float4 streaming)
// ---------------------------------------------------------------------------
__global__ __launch_bounds__(256) void gcn_self_init(
    const float* __restrict__ H, const float* __restrict__ sw,
    float* __restrict__ out, int n_f4)
{
    const int i4 = blockIdx.x * 256 + threadIdx.x;
    if (i4 < n_f4) {
        const int n = i4 >> 6;                 // 64 float4 per 256-ch row
        const float w = sw[n];
        F4 h = reinterpret_cast<const F4*>(H)[i4];
        F4 o;
#pragma unroll
        for (int j = 0; j < 4; ++j) o.v[j] = h.v[j] * w;
        reinterpret_cast<F4*>(out)[i4] = o;
    }
}

// ---------------------------------------------------------------------------
// Kernel 3: out[tar[e]] += edge_weight[e] * h[src[e]]  (L2-resident atomics)
// 64 threads per edge (float4 per thread), 4 edge slots x 4 iters per block.
// ---------------------------------------------------------------------------
__global__ __launch_bounds__(256) void gcn_edge_scatter(
    const float* __restrict__ H, const int* __restrict__ src,
    const int* __restrict__ tar, const float* __restrict__ ew,
    float* __restrict__ out, int nedges)
{
    const int c4   = threadIdx.x & 63;     // float4 group -> channels [4*c4, 4*c4+4)
    const int slot = threadIdx.x >> 6;     // 0..3
    const long ebase = (long)blockIdx.x * 16 + slot * 4;
#pragma unroll
    for (int j = 0; j < 4; ++j) {
        const long e = ebase + j;
        if (e < nedges) {
            const int   s = src[e];
            const int   t = tar[e];
            const float w = ew[e];
            F4 h = *reinterpret_cast<const F4*>(H + (size_t)s * OUT_CH + c4 * 4);
            float* op = out + (size_t)t * OUT_CH + c4 * 4;
            atomicAdd(op + 0, h.v[0] * w);
            atomicAdd(op + 1, h.v[1] * w);
            atomicAdd(op + 2, h.v[2] * w);
            atomicAdd(op + 3, h.v[3] * w);
        }
    }
}

extern "C" void kernel_launch(void* const* d_in, const int* in_sizes, int n_in,
                              void* d_out, int out_size, void* d_ws, size_t ws_size,
                              hipStream_t stream) {
    const float* x   = (const float*)d_in[0];
    const float* W   = (const float*)d_in[1];
    const float* b   = (const float*)d_in[2];
    const int*   src = (const int*)d_in[3];
    const int*   tar = (const int*)d_in[4];
    const float* ew  = (const float*)d_in[5];
    const float* sw  = (const float*)d_in[6];
    float* out = (float*)d_out;

    const int N  = in_sizes[6];            // 50000 nodes
    const int E  = in_sizes[3];            // 800000 edges
    const int NW = in_sizes[1];            // 256*1024 W elements

    // workspace layout: h [N,256] f32 | W_hi f16 | W_lo f16 (256B-aligned)
    char* ws = (char*)d_ws;
    float*    h   = (float*)ws;
    size_t    off = ((size_t)N * OUT_CH * sizeof(float) + 255) & ~(size_t)255;
    _Float16* Whi = (_Float16*)(ws + off);
    off += ((size_t)NW * sizeof(_Float16) + 255) & ~(size_t)255;
    _Float16* Wlo = (_Float16*)(ws + off);

    // Phase 0: split W once (tiny: 1 MB read)
    gcn_w_split<<<(NW / 4 + 255) / 256, 256, 0, stream>>>(W, Whi, Wlo, NW / 4);

    // Phase 1: h = x @ W^T + b   (WMMA)
    gcn_gemm_wmma<<<(N + 63) / 64, 512, 0, stream>>>(x, Whi, Wlo, b, h, N);

    // Phase 2: out = self_w * h
    const int n_f4 = (N * OUT_CH) / 4;
    gcn_self_init<<<(n_f4 + 255) / 256, 256, 0, stream>>>(h, sw, out, n_f4);

    // Phase 3: scatter-add edge messages
    gcn_edge_scatter<<<(E + 15) / 16, 256, 0, stream>>>(h, src, tar, ew, out, E);
}
